// MyNaiveRNN_27049704030528
// MI455X (gfx1250) — matrix operand order repaired and checked by reading, not compiled, not run on hardware
//
#include <hip/hip_runtime.h>
#include <math.h>

// Problem constants (from reference)
#define T_SEQ  16384
#define D_IN   512
#define D_HID  2048
#define D_OUT  512
#define LDWH   (D_HID + D_IN)   // 2560, row stride of Wh

#define CTB    4                // column tiles per block (shared via LDS)
#define WPB    8                // waves (row tiles) per block

typedef __attribute__((ext_vector_type(16))) __bf16 v16bf;
typedef __attribute__((ext_vector_type(8)))  float  v8f;

__device__ __forceinline__ unsigned short f2bf(float f) {
  union { float f; unsigned int u; } c; c.f = f;
  unsigned int u = c.u;
  unsigned int r = (u + 0x7FFFu + ((u >> 16) & 1u)) >> 16;  // RNE
  return (unsigned short)r;
}

// ---------------------------------------------------------------------------
// Pack fp32 matrix tiles into bf16 WMMA operand layouts (wave32, 16x16x32).
//   A layout (16x32, MxK): lane L holds row M=L%16; elem j -> K = 8*(L/16) + (j<8 ? j : j+8)
//   B layout (32x16, KxN): lane L holds col N=L%16; elem j -> K = 8*(L/16) + (j<8 ? j : j+8)
// dst flat index = (((rowTile*KC + kc)*32 + lane)*16 + j)
// ---------------------------------------------------------------------------
__global__ void pack_tiles(const float* __restrict__ src, unsigned short* __restrict__ dst,
                           int ldsrc, int koff, int kdim, int total, int isA)
{
  int idx = blockIdx.x * blockDim.x + threadIdx.x;
  if (idx >= total) return;
  int j    = idx & 15;
  int lane = (idx >> 4) & 31;
  int rest = idx >> 9;
  int KC   = kdim >> 5;
  int kc   = rest % KC;
  int rt   = rest / KC;
  int row  = rt * 16 + (lane & 15);
  int k;
  if (isA) k = kc * 32 + ((lane >> 4) << 3) + (j < 8 ? j : j + 8);
  else     k = kc * 32 + ((lane >> 4) << 3) + (j < 8 ? j : j + 8);
  dst[idx] = f2bf(src[(size_t)row * ldsrc + koff + k]);
}

// ---------------------------------------------------------------------------
// bf16 WMMA GEMM with LDS-staged B operand.
// Block = 8 waves: each wave owns one 16-row tile and CTB=4 column tiles
// (16x64 output strip, 4 fp32 accumulators). The 4 B tiles (4KB per k-chunk)
// are shared by all 8 waves, staged global->LDS with async-to-LDS copies
// (ASYNCcnt), double-buffered. A is loaded directly (32B/lane, contiguous).
// ---------------------------------------------------------------------------
__global__ void __launch_bounds__(256)
wmma_gemm_bf16(const unsigned short* __restrict__ Ap,
               const unsigned short* __restrict__ Bp,
               const float* __restrict__ bias,
               float* __restrict__ C,
               int colTiles, int KC, int ldc)
{
  __shared__ unsigned short Bs[2][CTB * 512];   // 2 x 4KB double buffer

  const int tid  = threadIdx.x;
  const int lane = tid & 31;
  const int wv   = tid >> 5;

  const int cBlocks = colTiles / CTB;
  const int cb  = blockIdx.x % cBlocks;
  const int rb  = blockIdx.x / cBlocks;
  const int rt  = rb * WPB + wv;
  const int ct0 = cb * CTB;

  // B staging: 4KB per k-chunk = 256 threads x 16B.
  //   tile bi = tid/64, chunk bq = tid%64 (16B granules within the 1KB tile)
  const int bi = tid >> 6;
  const int bq = tid & 63;
  const unsigned short* bsrc = Bp + ((size_t)(ct0 + bi) * KC) * 512 + bq * 8;
  const unsigned long long ldsA =
      (unsigned long long)(&Bs[0][bi * 512 + bq * 8]);
  const unsigned long long ldsB =
      (unsigned long long)(&Bs[1][bi * 512 + bq * 8]);
  const unsigned bofs[2] = { (unsigned)ldsA, (unsigned)ldsB };

  const v16bf* A = (const v16bf*)Ap + (size_t)rt * KC * 32 + lane;

  v8f c0 = {}, c1 = {}, c2 = {}, c3 = {};

  // prologue: stage k-chunk 0 into buffer 0
  {
    unsigned long long g = (unsigned long long)bsrc;
    asm volatile("global_load_async_to_lds_b128 %0, %1, off"
                 :: "v"(bofs[0]), "v"(g) : "memory");
  }

  for (int kc = 0; kc < KC; ++kc) {
    const int buf = kc & 1;
    if (kc + 1 < KC) {
      unsigned long long g = (unsigned long long)(bsrc + (size_t)(kc + 1) * 512);
      asm volatile("global_load_async_to_lds_b128 %0, %1, off"
                   :: "v"(bofs[buf ^ 1]), "v"(g) : "memory");
      asm volatile("s_wait_asynccnt 0x1" ::: "memory");
    } else {
      asm volatile("s_wait_asynccnt 0x0" ::: "memory");
    }
    __syncthreads();   // this wave's chunk-kc data is in LDS for everyone

    v16bf a = A[kc * 32];
    const v16bf* bb = (const v16bf*)&Bs[buf][0] + lane;  // lane's 32B of tile 0
    v16bf b0 = bb[0 * 32];
    v16bf b1 = bb[1 * 32];
    v16bf b2 = bb[2 * 32];
    v16bf b3 = bb[3 * 32];
    c0 = __builtin_amdgcn_wmma_f32_16x16x32_bf16(false, a, false, b0, (short)0, c0, false, false);
    c1 = __builtin_amdgcn_wmma_f32_16x16x32_bf16(false, a, false, b1, (short)0, c1, false, false);
    c2 = __builtin_amdgcn_wmma_f32_16x16x32_bf16(false, a, false, b2, (short)0, c2, false, false);
    c3 = __builtin_amdgcn_wmma_f32_16x16x32_bf16(false, a, false, b3, (short)0, c3, false, false);

    __syncthreads();   // everyone done reading before next overwrite
  }

  const int r0 = rt * 16 + ((lane >> 4) << 3);   // D: VGPR v -> row v + 8*(lane/16)
  const int nn = lane & 15;
  v8f cc[CTB] = { c0, c1, c2, c3 };
#pragma unroll
  for (int i = 0; i < CTB; ++i) {
    const int n = (ct0 + i) * 16 + nn;
    const float bv = bias[n];
    float* out = C + (size_t)r0 * ldc + n;
#pragma unroll
    for (int v = 0; v < 8; ++v)
      out[(size_t)v * ldc] = cc[i][v] + bv;
  }
}

// ---------------------------------------------------------------------------
// Init: copy h0 into ping buffer, reset device-wide barrier (ws is poisoned,
// and gen must return to 0 between graph replays).
// ---------------------------------------------------------------------------
__global__ void init_state(const float* __restrict__ h0, float* __restrict__ hPing,
                           int* __restrict__ bar)
{
  int i = blockIdx.x * blockDim.x + threadIdx.x;
  if (i < D_HID) hPing[i] = h0[i];
  if (i == 0) { bar[0] = 0; bar[1] = 0; }
}

// ---------------------------------------------------------------------------
// Sequential recurrence: h_{t+1} = tanh(Wh_h @ h_t + preX[t])  (bh in preX)
// Persistent 32-WG kernel; Wh_h (16MB fp32) lives in the 192MB L2.
// Each wave owns 8 output rows; h_t staged in LDS; wave32 shuffle reduction.
// Device-wide sync per step via atomic arrive + generation counter.
// Scatters bf16(h_t) directly into the packed-A layout of H for phase 3.
// ---------------------------------------------------------------------------
__global__ void rnn_steps(const float* __restrict__ Wh,      // [2048, 2560]
                          const float* __restrict__ preX,    // [T, 2048]
                          float* __restrict__ hPing, float* __restrict__ hPong,
                          unsigned short* __restrict__ Hp,   // packed A tiles of H
                          int* __restrict__ bar, int nwg)
{
  __shared__ float hs[D_HID];
  const int tid     = threadIdx.x;
  const int lane    = tid & 31;
  const int wv      = tid >> 5;                       // 8 waves / WG
  const int rowBase = blockIdx.x * 64 + wv * 8;       // 64 rows per WG
  const int KC      = D_HID >> 5;                     // k-chunks per row of Hp

  for (int t = 0; t < T_SEQ; ++t) {
    const float* hcur  = (t & 1) ? hPong : hPing;
    float*       hnext = (t & 1) ? hPing : hPong;

    for (int i = tid; i < D_HID; i += blockDim.x) hs[i] = hcur[i];
    __syncthreads();

    const float* px = preX + (size_t)t * D_HID;
#pragma unroll 2
    for (int rr = 0; rr < 8; ++rr) {
      const int n = rowBase + rr;
      const float* wrow = Wh + (size_t)n * LDWH + D_IN;   // Wh_h row n
      // prefetch next row's weights (lane-strided, 256B apart -> whole 8KB row)
      if (rr < 7)
        __builtin_prefetch(wrow + LDWH + lane * 64, 0, 1);
      float acc = 0.f;
#pragma unroll 8
      for (int k = lane; k < D_HID; k += 32)
        acc = fmaf(wrow[k], hs[k], acc);
      for (int off = 16; off > 0; off >>= 1)
        acc += __shfl_xor(acc, off, 32);
      if (lane == 0) {
        const float hn = tanhf(acc + px[n]);
        hnext[n] = hn;
        // scatter into packed-A layout: row t, column n
        const int tt  = t >> 4, m = t & 15;
        const int kc  = n >> 5, r = n & 31;
        const int hlf = (r >> 3) & 1;
        const int j   = (r & 7) + ((r & 16) >> 1);
        const int pl  = m + (hlf << 4);
        Hp[((((size_t)tt * KC + kc) * 32 + pl) << 4) + j] = f2bf(hn);
      }
    }

    // ---- device-wide barrier (arrive + generation) ----
    __threadfence();
    __syncthreads();
    if (tid == 0) {
      int prev = __hip_atomic_fetch_add(&bar[0], 1, __ATOMIC_ACQ_REL,
                                        __HIP_MEMORY_SCOPE_AGENT);
      if (prev == nwg - 1) {
        __hip_atomic_store(&bar[0], 0, __ATOMIC_RELAXED, __HIP_MEMORY_SCOPE_AGENT);
        __hip_atomic_fetch_add(&bar[1], 1, __ATOMIC_RELEASE,
                               __HIP_MEMORY_SCOPE_AGENT);
      } else {
        while (__hip_atomic_load(&bar[1], __ATOMIC_ACQUIRE,
                                 __HIP_MEMORY_SCOPE_AGENT) <= t) {
          __builtin_amdgcn_s_sleep(1);
        }
      }
    }
    __syncthreads();
  }
}

// ---------------------------------------------------------------------------
extern "C" void kernel_launch(void* const* d_in, const int* in_sizes, int n_in,
                              void* d_out, int out_size, void* d_ws, size_t ws_size,
                              hipStream_t stream)
{
  (void)in_sizes; (void)n_in; (void)out_size; (void)ws_size;
  const float* X  = (const float*)d_in[0];   // [16384, 512]
  const float* h0 = (const float*)d_in[1];   // [2048]
  const float* Wh = (const float*)d_in[2];   // [2048, 2560]
  const float* bh = (const float*)d_in[3];   // [2048]
  const float* Wo = (const float*)d_in[4];   // [512, 2048]
  const float* bo = (const float*)d_in[5];   // [512]
  float* Y = (float*)d_out;                  // [16384, 512]

  char* ws = (char*)d_ws;
  size_t off = 0;
  auto alloc = [&](size_t bytes) -> void* {
    void* p = ws + off;
    off = (off + bytes + 255) & ~(size_t)255;
    return p;
  };

  int*            bar   = (int*)alloc(256);
  float*          hPing = (float*)alloc((size_t)D_HID * 4);
  float*          hPong = (float*)alloc((size_t)D_HID * 4);
  unsigned short* Xp    = (unsigned short*)alloc((size_t)T_SEQ * D_IN * 2);
  unsigned short* Whxp  = (unsigned short*)alloc((size_t)D_HID * D_IN * 2);
  unsigned short* Wop   = (unsigned short*)alloc((size_t)D_OUT * D_HID * 2);
  float*          preX  = (float*)alloc((size_t)T_SEQ * D_HID * 4);
  unsigned short* Hp    = (unsigned short*)alloc((size_t)T_SEQ * D_HID * 2);

  init_state<<<(D_HID + 255) / 256, 256, 0, stream>>>(h0, hPing, bar);

  // Pack operands into WMMA lane layouts (bf16).
  pack_tiles<<<(T_SEQ * D_IN) / 256, 256, 0, stream>>>(X, Xp, D_IN, 0, D_IN,
                                                       T_SEQ * D_IN, 1);
  pack_tiles<<<(D_HID * D_IN) / 256, 256, 0, stream>>>(Wh, Whxp, LDWH, 0, D_IN,
                                                       D_HID * D_IN, 0);
  pack_tiles<<<(D_OUT * D_HID) / 256, 256, 0, stream>>>(Wo, Wop, D_HID, 0, D_HID,
                                                        D_OUT * D_HID, 0);

  // Phase 1: preX = X @ Wh_x^T + bh.  128 row-blocks x 32 col-blocks.
  wmma_gemm_bf16<<<dim3((1024 / WPB) * (128 / CTB)), 256, 0, stream>>>(
      Xp, Whxp, bh, preX, 128, D_IN / 32, D_HID);

  // Phase 2: sequential tanh recurrence (persistent grid, 32 WGs).
  rnn_steps<<<32, 256, 0, stream>>>(Wh, preX, hPing, hPong, Hp, bar, 32);

  // Phase 3: Y = H @ Wo^T + bo.  128 row-blocks x 8 col-blocks.
  wmma_gemm_bf16<<<dim3((1024 / WPB) * (32 / CTB)), 256, 0, stream>>>(
      Hp, Wop, bo, Y, 32, D_HID / 32, D_OUT);
}
